// GPTCustom_68556267979204
// MI455X (gfx1250) — matrix operand order repaired and checked by reading, not compiled, not run on hardware
//
#include <hip/hip_runtime.h>
#include <hip/hip_bf16.h>
#include <type_traits>

// ---------------------------------------------------------------------------
// GPT forward for MI455X (gfx1250, wave32).
// bf16 weight cache + bf16 shadow activations -> every GEMM is bf16 x bf16
// with f32 accumulation via v_wmma_f32_16x16x32_bf16.  Tiles are streamed
// into double-buffered LDS with GLOBAL_LOAD_ASYNC_TO_LDS_B128 (ASYNCcnt),
// so the copy engine runs decoupled from the WMMA stream.
// ---------------------------------------------------------------------------

typedef __attribute__((ext_vector_type(16))) __bf16 v16bf;
typedef __attribute__((ext_vector_type(8)))  float  v8f;
typedef __attribute__((ext_vector_type(4)))  float  f32x4;
typedef __attribute__((ext_vector_type(4)))  unsigned int u32x4;
typedef __attribute__((ext_vector_type(4)))  int v4i;
typedef __attribute__((address_space(1))) v4i* gptr_v4i;   // global int4*
typedef __attribute__((address_space(3))) v4i* lptr_v4i;   // LDS int4*

union AFrag { v16bf v; u32x4 u[2]; };
union Pack8 { __bf16 h[8]; u32x4 u; };
union Pack4 { __bf16 h[4]; unsigned int u[2]; };

// Model dims (fixed by the reference)
enum { GB = 2, GS = 1024, GD = 1024, GH = 16, GHD = 64, GF = 4096, GV = 32000, GL = 4 };

// GEMM tile config: 8 waves (256 threads), tile BM x BNT x BK, BNT in {64,128}
enum { BM = 128, BK = 64, NTHREADS = 256, LDSPAD = 8, LDW = BK + LDSPAD };

__device__ __forceinline__ float gelu_exact(float x) {
    return 0.5f * x * (1.0f + erff(x * 0.70710678118654752f));
}

// ---------------------------------------------------------------------------
// Async global -> LDS 16-byte copy (per lane), tracked by ASYNCcnt.
// ---------------------------------------------------------------------------
__device__ __forceinline__ void async_cp16(const __bf16* g, __bf16* l) {
#if __has_builtin(__builtin_amdgcn_global_load_async_to_lds_b128)
    __builtin_amdgcn_global_load_async_to_lds_b128(
        (gptr_v4i)(unsigned long long)g,
        (lptr_v4i)(unsigned)(unsigned long long)l,   // low 32 bits = LDS offset
        0, 0);
#else
    unsigned lo = (unsigned)(unsigned long long)l;   // LDS aperture offset
    unsigned long long ga = (unsigned long long)g;
    asm volatile("global_load_async_to_lds_b128 %0, %1, off"
                 :: "v"(lo), "v"(ga) : "memory");
#endif
}

__device__ __forceinline__ void wait_async0() {
#if __has_builtin(__builtin_amdgcn_s_wait_asynccnt)
    __builtin_amdgcn_s_wait_asynccnt(0);
#else
    asm volatile("s_wait_asynccnt 0" ::: "memory");
#endif
}

// ---------------------------------------------------------------------------
// Issue the async copies for one K-tile (pure 16B per-lane transfers).
// ---------------------------------------------------------------------------
template <int BNT>
__device__ __forceinline__ void stage_async(const __bf16* __restrict__ A,
                                            const __bf16* __restrict__ B,
                                            int tid, int m0, int n0, int k0,
                                            int lda, int ldb,
                                            __bf16 (*As)[LDW], __bf16 (*Bs)[LDW]) {
    #pragma unroll
    for (int cc = 0; cc < (BM * BK / 8) / NTHREADS; ++cc) {   // 4
        int ci  = tid + cc * NTHREADS;
        int row = ci >> 3;
        int c8  = (ci & 7) << 3;
        async_cp16(A + (size_t)(m0 + row) * lda + k0 + c8, &As[row][c8]);
    }
    #pragma unroll
    for (int cc = 0; cc < (BNT * BK / 8) / NTHREADS; ++cc) {  // 2 or 4
        int ci  = tid + cc * NTHREADS;
        int row = ci >> 3;
        int c8  = (ci & 7) << 3;
        async_cp16(B + (size_t)(n0 + row) * ldb + k0 + c8, &Bs[row][c8]);
    }
}

// ---------------------------------------------------------------------------
// C[M,N] = act( A[M,K](bf16) @ B[N,K](bf16)^T + bias(f32) )
// OUTMODE: 0 = f32 C ; 1 = bf16 C ; 2 = bf16 C stored transposed (C[n][m],
//          packed 16B stores -- 8 consecutive m per lane are contiguous).
// Batched via blockIdx.z: z -> (b = z/batchH, h = z%batchH) element strides.
// Requires: M % 128 == 0, N % BNT == 0, K % 64 == 0 (true for every call).
// ---------------------------------------------------------------------------
template <int BNT, bool DO_GELU, int OUTMODE>
__global__ __launch_bounds__(NTHREADS)
void gemm_wmma_bf16(const __bf16* __restrict__ A, const __bf16* __restrict__ B,
                    const float* __restrict__ bias, void* __restrict__ Cv,
                    int M, int N, int K, int lda, int ldb, int ldc,
                    int batchH,
                    long long aSB, long long aSH,
                    long long bSB, long long bSH,
                    long long cSB, long long cSH) {
    constexpr int FN = BNT / 32;           // B fragments per wave (2 or 4)
    __shared__ __attribute__((aligned(16))) __bf16 As[2][BM][LDW];
    __shared__ __attribute__((aligned(16))) __bf16 Bs[2][BNT][LDW];

    const int z  = blockIdx.z;
    const int bb = z / batchH;
    const int hh = z % batchH;
    A += (size_t)bb * aSB + (size_t)hh * aSH;
    B += (size_t)bb * bSB + (size_t)hh * bSH;
    using CT = typename std::conditional<OUTMODE == 0, float, __bf16>::type;
    CT* C = (CT*)Cv + (size_t)bb * cSB + (size_t)hh * cSH;

    const int tid  = threadIdx.x;
    const int lane = tid & 31;
    const int wid  = tid >> 5;
    const int rm   = (wid & 3) * 32;           // 4 waves along M
    const int cn   = (wid >> 2) * (BNT / 2);   // 2 waves along N

    const int m0 = blockIdx.y * BM;
    const int n0 = blockIdx.x * BNT;

    const int laneM = lane >> 4;               // 0/1 : K/M half selector
    const int laneL = lane & 15;

    v8f acc[2][FN] = {};

    const int T = K / BK;
    stage_async<BNT>(A, B, tid, m0, n0, 0, lda, ldb, As[0], Bs[0]);

    for (int t = 0; t < T; ++t) {
        wait_async0();                 // my tile-t copies have landed in LDS
        __syncthreads();               // everyone's have
        const int buf = t & 1;

        if (t + 1 < T)                 // tile t+1 DMA flies under the WMMAs;
            stage_async<BNT>(A, B, tid, m0, n0, (t + 1) * BK, lda, ldb,
                             As[buf ^ 1], Bs[buf ^ 1]);   // buffer last read at t-1

        #pragma unroll
        for (int ko = 0; ko < BK; ko += 32) {
            AFrag afr[2], bfr[FN];
            #pragma unroll
            for (int tm = 0; tm < 2; ++tm) {
                // A 16x32 bf16: m = lane&15; lanes>=16 take K halves 8..15 / 24..31
                const int am = rm + tm * 16 + laneL;
                afr[tm].u[0] = *(const u32x4*)(&As[buf][am][ko + 8 * laneM]);
                afr[tm].u[1] = *(const u32x4*)(&As[buf][am][ko + 16 + 8 * laneM]);
            }
            #pragma unroll
            for (int tn = 0; tn < FN; ++tn) {
                // B 32x16 bf16: n = lane&15; lanes>=16 take K halves 8..15 / 24..31
                const int bn = cn + tn * 16 + laneL;
                bfr[tn].u[0] = *(const u32x4*)(&Bs[buf][bn][ko + 8 * laneM]);
                bfr[tn].u[1] = *(const u32x4*)(&Bs[buf][bn][ko + 16 + 8 * laneM]);
            }
            #pragma unroll
            for (int tm = 0; tm < 2; ++tm)
                #pragma unroll
                for (int tn = 0; tn < FN; ++tn)
                    acc[tm][tn] = __builtin_amdgcn_wmma_f32_16x16x32_bf16(
                        false, afr[tm].v, false, bfr[tn].v,
                        (short)0, acc[tm][tn], false, false);
        }
    }

    // ---- epilogue: C/D layout: m = r + 8*(lane>>4), n = lane&15 per VGPR r ----
    #pragma unroll
    for (int tm = 0; tm < 2; ++tm) {
        #pragma unroll
        for (int tn = 0; tn < FN; ++tn) {
            const int gm = m0 + rm + tm * 16 + laneM * 8;
            const int gn = n0 + cn + tn * 16 + laneL;
            const float bv = bias ? bias[gn] : 0.0f;
            if constexpr (OUTMODE == 2) {
                Pack8 pk;
                #pragma unroll
                for (int r = 0; r < 8; ++r) {
                    float x = acc[tm][tn][r] + bv;
                    if (DO_GELU) x = gelu_exact(x);
                    pk.h[r] = (__bf16)x;
                }
                *(u32x4*)(&C[(size_t)gn * ldc + gm]) = pk.u;   // 16B transposed store
            } else {
                #pragma unroll
                for (int r = 0; r < 8; ++r) {
                    float x = acc[tm][tn][r] + bv;
                    if (DO_GELU) x = gelu_exact(x);
                    C[(size_t)(gm + r) * ldc + gn] = (CT)x;
                }
            }
        }
    }
}

// ---------------------------------------------------------------------------
// fp32 -> bf16 (4-wide), n % 4 == 0
// ---------------------------------------------------------------------------
__global__ void cvt_bf16_kernel(const float* __restrict__ src, __bf16* __restrict__ dst,
                                long long n4) {
    for (long long i = (long long)blockIdx.x * blockDim.x + threadIdx.x; i < n4;
         i += (long long)gridDim.x * blockDim.x) {
        const f32x4 f = *(const f32x4*)(src + i * 4);
        Pack4 pk;
        #pragma unroll
        for (int j = 0; j < 4; ++j) pk.h[j] = (__bf16)f[j];
        *(unsigned long long*)(dst + i * 4) = *(unsigned long long*)pk.u;
    }
}

// ---------------------------------------------------------------------------
// h[b,s,:] = tok_emb[x[b,s]] + pos_emb[s]   (f32 + bf16 shadow)
// ---------------------------------------------------------------------------
__global__ void embed_kernel(const int* __restrict__ x, const float* __restrict__ tok,
                             const float* __restrict__ pos, float* __restrict__ h,
                             __bf16* __restrict__ hb, long long total) {
    for (long long i = (long long)blockIdx.x * blockDim.x + threadIdx.x; i < total;
         i += (long long)gridDim.x * blockDim.x) {
        long long bs = i / GD;
        int d = (int)(i % GD);
        int s = (int)(bs % GS);
        int t = x[bs];
        float v = tok[(size_t)t * GD + d] + pos[(size_t)s * GD + d];
        h[i]  = v;
        hb[i] = (__bf16)v;
    }
}

// ---------------------------------------------------------------------------
// softmax over rows: reads f32 scores, writes bf16 attention weights
// ---------------------------------------------------------------------------
__global__ __launch_bounds__(256) void softmax_rows(const float* __restrict__ e,
                                                    __bf16* __restrict__ eb,
                                                    int S, float scale) {
    const float* row = e + (size_t)blockIdx.x * S;
    __bf16* rowo = eb + (size_t)blockIdx.x * S;
    __shared__ float red[256];
    const int t = threadIdx.x;

    float m = -3.4e38f;
    for (int i = t; i < S; i += 256) m = fmaxf(m, row[i] * scale);
    red[t] = m; __syncthreads();
    for (int s2 = 128; s2 > 0; s2 >>= 1) {
        if (t < s2) red[t] = fmaxf(red[t], red[t + s2]);
        __syncthreads();
    }
    const float rmax = red[0]; __syncthreads();

    float vloc[8];
    float sum = 0.0f;
    int nj = 0;
    for (int i = t; i < S; i += 256, ++nj) {
        float v = __expf(row[i] * scale - rmax);
        vloc[nj] = v;
        sum += v;
    }
    red[t] = sum; __syncthreads();
    for (int s2 = 128; s2 > 0; s2 >>= 1) {
        if (t < s2) red[t] += red[t + s2];
        __syncthreads();
    }
    const float inv = 1.0f / red[0];
    nj = 0;
    for (int i = t; i < S; i += 256, ++nj)
        rowo[i] = (__bf16)(vloc[nj] * inv);
}

// ---------------------------------------------------------------------------
// out = LayerNorm(a + b) * g + be   (f32 + bf16 shadow; one block per row)
// ---------------------------------------------------------------------------
__global__ __launch_bounds__(256) void add_layernorm(const float* __restrict__ a,
                                                     const float* __restrict__ b,
                                                     const float* __restrict__ g,
                                                     const float* __restrict__ be,
                                                     float* __restrict__ out,
                                                     __bf16* __restrict__ outb, int D) {
    const size_t row = blockIdx.x;
    const float* pa = a + row * D;
    const float* pb = b + row * D;
    float* po = out + row * D;
    __bf16* pob = outb + row * D;
    __shared__ float red[256];
    const int t = threadIdx.x;

    float vloc[8];
    float sum = 0.0f;
    int nj = 0;
    for (int i = t; i < D; i += 256, ++nj) {
        float v = pa[i] + pb[i];
        vloc[nj] = v;
        sum += v;
    }
    red[t] = sum; __syncthreads();
    for (int s2 = 128; s2 > 0; s2 >>= 1) {
        if (t < s2) red[t] += red[t + s2];
        __syncthreads();
    }
    const float mu = red[0] / (float)D; __syncthreads();

    float sum2 = 0.0f;
    for (int j = 0; j < nj; ++j) { float d = vloc[j] - mu; sum2 += d * d; }
    red[t] = sum2; __syncthreads();
    for (int s2 = 128; s2 > 0; s2 >>= 1) {
        if (t < s2) red[t] += red[t + s2];
        __syncthreads();
    }
    const float rstd = rsqrtf(red[0] / (float)D + 1e-5f);

    nj = 0;
    for (int i = t; i < D; i += 256, ++nj) {
        float y = (vloc[nj] - mu) * rstd * g[i] + be[i];
        po[i]  = y;
        pob[i] = (__bf16)y;
    }
}

// ---------------------------------------------------------------------------
// host-side launcher helpers
// ---------------------------------------------------------------------------
template <int BNT, bool G, int OM>
static void rungemm(hipStream_t st, const __bf16* A, const __bf16* B, const float* bias,
                    void* C, int M, int N, int K, int lda, int ldb, int ldc,
                    int batches = 1, int batchH = 1,
                    long long aSB = 0, long long aSH = 0,
                    long long bSB = 0, long long bSH = 0,
                    long long cSB = 0, long long cSH = 0) {
    dim3 grid(N / BNT, M / BM, batches);
    gemm_wmma_bf16<BNT, G, OM><<<grid, dim3(NTHREADS), 0, st>>>(
        A, B, bias, C, M, N, K, lda, ldb, ldc, batchH, aSB, aSH, bSB, bSH, cSB, cSH);
}

static void cvt(hipStream_t st, const float* src, __bf16* dst, long long n) {
    long long n4 = n / 4;
    int blocks = (int)((n4 + 255) / 256);
    if (blocks > 16384) blocks = 16384;
    cvt_bf16_kernel<<<blocks, 256, 0, st>>>(src, dst, n4);
}

extern "C" void kernel_launch(void* const* d_in, const int* in_sizes, int n_in,
                              void* d_out, int out_size, void* d_ws, size_t ws_size,
                              hipStream_t stream) {
    (void)in_sizes; (void)n_in; (void)out_size; (void)ws_size;

    const int*   x       = (const int*)d_in[0];
    const float* tok_emb = (const float*)d_in[1];
    const float* pos_emb = (const float*)d_in[2];
    const float* wq      = (const float*)d_in[3];
    const float* wk      = (const float*)d_in[4];
    const float* wv      = (const float*)d_in[5];
    const float* wo      = (const float*)d_in[6];
    const float* bo      = (const float*)d_in[7];
    const float* ln1_g   = (const float*)d_in[8];
    const float* ln1_b   = (const float*)d_in[9];
    const float* ln2_g   = (const float*)d_in[10];
    const float* ln2_b   = (const float*)d_in[11];
    const float* w1      = (const float*)d_in[12];
    const float* b1      = (const float*)d_in[13];
    const float* w2      = (const float*)d_in[14];
    const float* b2      = (const float*)d_in[15];
    const float* out_w   = (const float*)d_in[16];
    const float* out_b   = (const float*)d_in[17];
    float* logits = (float*)d_out;

    // ---- workspace layout ----
    const size_t nBSD = (size_t)GB * GS * GD;        // 2M elements
    const size_t nBSF = (size_t)GB * GS * GF;        // 8M
    const size_t nE   = (size_t)GB * GH * GS * GS;   // 33.5M
    char* cur = (char*)d_ws;
    auto falloc = [&](size_t n) { float*  p = (float*)cur;  cur += n * sizeof(float);  return p; };
    auto halloc = [&](size_t n) { __bf16* p = (__bf16*)cur; cur += n * sizeof(__bf16); return p; };

    float*  h    = falloc(nBSD);   // f32 activations (residual stream)
    float*  op   = falloc(nBSD);
    float*  x1   = falloc(nBSD);
    float*  ff2  = falloc(nBSD);
    float*  e    = falloc(nE);     // f32 attention scores
    __bf16* hb   = halloc(nBSD);   // bf16 shadows / GEMM operands
    __bf16* q    = halloc(nBSD);
    __bf16* kb   = halloc(nBSD);
    __bf16* vT   = halloc(nBSD);   // [B,H,64,S]
    __bf16* oh   = halloc(nBSD);
    __bf16* x1b  = halloc(nBSD);
    __bf16* ff1b = halloc(nBSF);
    __bf16* eb   = halloc(nE);     // bf16 attention weights
    __bf16* wq_b = halloc((size_t)GL * GHD * GHD);
    __bf16* wk_b = halloc((size_t)GL * GHD * GHD);
    __bf16* wv_b = halloc((size_t)GL * GHD * GHD);
    __bf16* wo_b = halloc((size_t)GL * GD * GD);
    __bf16* w1_b = halloc((size_t)GL * GF * GD);
    __bf16* w2_b = halloc((size_t)GL * GD * GF);
    __bf16* ow_b = halloc((size_t)GV * GD);

    const int BS = GB * GS;        // 2048 token rows
    const int MH = BS * GH;        // 32768 head rows

    // ---- one-time weight conversions (per launch; deterministic) ----
    cvt(stream, wq, wq_b, (long long)GL * GHD * GHD);
    cvt(stream, wk, wk_b, (long long)GL * GHD * GHD);
    cvt(stream, wv, wv_b, (long long)GL * GHD * GHD);
    cvt(stream, wo, wo_b, (long long)GL * GD * GD);
    cvt(stream, w1, w1_b, (long long)GL * GF * GD);
    cvt(stream, w2, w2_b, (long long)GL * GD * GF);
    cvt(stream, out_w, ow_b, (long long)GV * GD);

    // ---- embeddings ----
    embed_kernel<<<(int)(nBSD / 256), 256, 0, stream>>>(x, tok_emb, pos_emb, h, hb,
                                                        (long long)nBSD);

    const float inv_scale = 1.0f / 32.0f;  // 1/sqrt(D=1024), per reference

    for (int l = 0; l < GL; ++l) {
        const __bf16* wq_l = wq_b + (size_t)l * GHD * GHD;
        const __bf16* wk_l = wk_b + (size_t)l * GHD * GHD;
        const __bf16* wv_l = wv_b + (size_t)l * GHD * GHD;
        const __bf16* wo_l = wo_b + (size_t)l * GD * GD;
        const __bf16* w1_l = w1_b + (size_t)l * GF * GD;
        const __bf16* w2_l = w2_b + (size_t)l * GD * GF;
        const float* bo_l  = bo    + (size_t)l * GD;
        const float* g1_l  = ln1_g + (size_t)l * GD;
        const float* be1_l = ln1_b + (size_t)l * GD;
        const float* g2_l  = ln2_g + (size_t)l * GD;
        const float* be2_l = ln2_b + (size_t)l * GD;
        const float* b1_l  = b1 + (size_t)l * GF;
        const float* b2_l  = b2 + (size_t)l * GD;

        // Q,K projections: [B*S*H,64] @ W[64,64]^T -> bf16
        rungemm<64, false, 1>(stream, hb, wq_l, nullptr, q,  MH, GHD, GHD, GHD, GHD, GHD);
        rungemm<64, false, 1>(stream, hb, wk_l, nullptr, kb, MH, GHD, GHD, GHD, GHD, GHD);
        // V projection, batched per (b,h), stored transposed: vT[b,h,d,s]
        rungemm<64, false, 2>(stream, hb, wv_l, nullptr, vT, GS, GHD, GHD, GD, GHD, GS,
                              GB * GH, GH,
                              (long long)GS * GD, GHD, 0, 0,
                              (long long)GH * GHD * GS, (long long)GHD * GS);

        // scores: per (b,h): Q[S,64] @ K[S,64]^T -> e[b,h,S,S] (f32)
        rungemm<128, false, 0>(stream, q, kb, nullptr, e, GS, GS, GHD, GD, GD, GS,
                               GB * GH, GH,
                               (long long)GS * GD, GHD,
                               (long long)GS * GD, GHD,
                               (long long)GH * GS * GS, (long long)GS * GS);

        softmax_rows<<<GB * GH * GS, 256, 0, stream>>>(e, eb, GS, inv_scale);

        // attn @ V: per (b,h): eb[S,S] @ vT[64,S]^T -> oh[b,:,h,:] (bf16)
        rungemm<64, false, 1>(stream, eb, vT, nullptr, oh, GS, GHD, GS, GS, GS, GD,
                              GB * GH, GH,
                              (long long)GH * GS * GS, (long long)GS * GS,
                              (long long)GH * GHD * GS, (long long)GHD * GS,
                              (long long)GS * GD, GHD);

        // out projection + bias (f32 for residual)
        rungemm<128, false, 0>(stream, oh, wo_l, bo_l, op, BS, GD, GD, GD, GD, GD);

        // x1 = LN(op + h)  (f32 + bf16 shadow)
        add_layernorm<<<BS, 256, 0, stream>>>(op, h, g1_l, be1_l, x1, x1b, GD);

        // FFN: gelu(x1 @ w1^T + b1) -> bf16 ; @ w2^T + b2 -> f32
        rungemm<128, true, 1>(stream, x1b, w1_l, b1_l, ff1b, BS, GF, GD, GD, GD, GF);
        rungemm<128, false, 0>(stream, ff1b, w2_l, b2_l, ff2, BS, GD, GF, GF, GF, GD);

        // h = LN(ff2 + x1)  (f32 + bf16 shadow for next layer / logits)
        add_layernorm<<<BS, 256, 0, stream>>>(ff2, x1, g2_l, be2_l, h, hb, GD);
    }

    // logits = hb @ out_w^T + out_b  (M=2048, N=32000, K=1024)
    rungemm<128, false, 0>(stream, hb, ow_b, out_b, logits, BS, GV, GD, GD, GD, GV);
}
